// EFNHybrid_76441827934550
// MI455X (gfx1250) — matrix operand order repaired and checked by reading, not compiled, not run on hardware
//
#include <hip/hip_runtime.h>

#define N_NODES 51200
#define N_EDGES 819200
#define K1_WAVES 8
#define K3_WAVES 4

// Padded LDS row strides (elements): +8 bf16 = +16B so that row stride mod
// 64 banks = 4 words -> conflict-free ds_load_b128 across each 16-lane phase.
#define S128 136   // for 128-wide rows (272B, 16B-aligned)
#define S256 264   // for 256-wide rows (528B, 16B-aligned)

typedef __bf16 bf16_t;
typedef __attribute__((ext_vector_type(16))) __bf16 bf16x16;
typedef __attribute__((ext_vector_type(8)))  float  f32x8;
typedef __attribute__((ext_vector_type(4)))  unsigned int u32x4;

union Frag { bf16x16 v; u32x4 q[2]; };

static __device__ __forceinline__ unsigned int pk_bf16(float a, float b) {
  unsigned short ua = __builtin_bit_cast(unsigned short, (bf16_t)a);
  unsigned short ub = __builtin_bit_cast(unsigned short, (bf16_t)b);
  return (unsigned int)ua | ((unsigned int)ub << 16);
}
static __device__ __forceinline__ unsigned short bf16_bits(float a) {
  return __builtin_bit_cast(unsigned short, (bf16_t)a);
}
// Load a 16xbf16 WMMA fragment from LDS: two 16B chunks, second at +gap elems.
// A-fragment: gap=8  elems (+16B, contiguous K 16h..16h+15 per lane half, same as B).
// B-fragment: gap=8  elems (+16B, contiguous K within column-major weights).
static __device__ __forceinline__ bf16x16 ld_frag(const unsigned short* p, int gap) {
  Frag f;
  f.q[0] = *(const u32x4*)(p);
  f.q[1] = *(const u32x4*)(p + gap);
  return f.v;
}
static __device__ __forceinline__ void wave_lds_fence() {
  __asm__ volatile("s_wait_dscnt 0x0" ::: "memory");
}

// ---------------- zero-fill workspace/output ----------------
__global__ void efn_zero(float* __restrict__ g, float* __restrict__ pooled,
                         float* __restrict__ out) {
  long idx = (long)blockIdx.x * blockDim.x + threadIdx.x;
  const long ng = (long)N_NODES * 64, np = 64 * 64, no = N_NODES;
  if (idx < ng)                 g[idx] = 0.0f;
  else if (idx < ng + np)       pooled[idx - ng] = 0.0f;
  else if (idx < ng + np + no)  out[idx - ng - np] = 0.0f;
}

// ---------------- kernel 1: global PTConv (edge MLP 128->128->64, scatter) ----------------
__global__ __launch_bounds__(256) void efn_global_conv(
    const float* __restrict__ x, const int* __restrict__ eidx,
    const float* __restrict__ gW1, const float* __restrict__ gb1,
    const float* __restrict__ gW2, const float* __restrict__ gb2,
    float* __restrict__ g)
{
  __shared__ __align__(16) unsigned short w1c[128 * S128]; // gW1 col-major [n][k] bf16
  __shared__ __align__(16) unsigned short w2c[64 * S128];  // gW2 col-major [n][k] bf16
  __shared__ float b1s[128];
  __shared__ float b2s[64];
  __shared__ __align__(16) unsigned short Ast[K1_WAVES][16 * S128]; // edge rows bf16
  __shared__ __align__(16) unsigned short Hst[K1_WAVES][16 * S128]; // hidden rows bf16
  __shared__ int dstI[K1_WAVES][16];

  const int tid = threadIdx.x;
  for (int i = tid; i < 128 * 128; i += 256) { int k = i >> 7, n = i & 127; w1c[n * S128 + k] = bf16_bits(gW1[i]); }
  for (int i = tid; i < 64 * 128;  i += 256) { int k = i >> 6, n = i & 63;  w2c[n * S128 + k] = bf16_bits(gW2[i]); }
  if (tid < 128) b1s[tid] = gb1[tid];
  if (tid < 64)  b2s[tid] = gb2[tid];
  __syncthreads();

  const int wave = tid >> 5;
  const int lane = tid & 31;
  const int m    = lane & 15;   // row (edge) for A, column for B/C
  const int half = lane >> 4;   // K-half / M-half selector
  const int tile = blockIdx.x * K1_WAVES + wave;

  unsigned short* Aw = Ast[wave];
  unsigned short* Hw = Hst[wave];

  // gather: row m = [x[dst] | x[src]] -> bf16
  {
    const int e  = tile * 16 + m;
    const int nd = (half == 0) ? eidx[N_EDGES + e] : eidx[e]; // dst then src halves
    if (half == 0) dstI[wave][m] = nd;
    const float4* xr = (const float4*)(x + (long)nd * 64);
    unsigned short* dp = Aw + m * S128 + half * 64;
#pragma unroll
    for (int j = 0; j < 16; j++) {
      float4 v = xr[j];
      *(uint2*)(dp + j * 4) = make_uint2(pk_bf16(v.x, v.y), pk_bf16(v.z, v.w));
    }
  }
  wave_lds_fence();

  // GEMM1: [16x128] @ [128x128] + b1, ReLU
  bf16x16 a0 = ld_frag(Aw + m * S128 +  0 + half * 16, 8);
  bf16x16 a1 = ld_frag(Aw + m * S128 + 32 + half * 16, 8);
  bf16x16 a2 = ld_frag(Aw + m * S128 + 64 + half * 16, 8);
  bf16x16 a3 = ld_frag(Aw + m * S128 + 96 + half * 16, 8);

#pragma unroll
  for (int nt = 0; nt < 8; nt++) {
    float bias = b1s[nt * 16 + m];
    f32x8 acc = {bias, bias, bias, bias, bias, bias, bias, bias};
    const unsigned short* bp = w1c + (nt * 16 + m) * S128 + half * 16;
    acc = __builtin_amdgcn_wmma_f32_16x16x32_bf16(false, a0, false, ld_frag(bp +  0, 8), (short)0, acc, false, false);
    acc = __builtin_amdgcn_wmma_f32_16x16x32_bf16(false, a1, false, ld_frag(bp + 32, 8), (short)0, acc, false, false);
    acc = __builtin_amdgcn_wmma_f32_16x16x32_bf16(false, a2, false, ld_frag(bp + 64, 8), (short)0, acc, false, false);
    acc = __builtin_amdgcn_wmma_f32_16x16x32_bf16(false, a3, false, ld_frag(bp + 96, 8), (short)0, acc, false, false);
#pragma unroll
    for (int r = 0; r < 8; r++) { // D layout: lane holds col n=nt*16+m, rows half*8+r
      Hw[(half * 8 + r) * S128 + nt * 16 + m] = bf16_bits(fmaxf(acc[r], 0.0f));
    }
  }
  wave_lds_fence();

  // GEMM2: [16x128] @ [128x64] + b2, scatter-add by dst
  int dloc[8];
#pragma unroll
  for (int r = 0; r < 8; r++) dloc[r] = dstI[wave][half * 8 + r];

  bf16x16 h0 = ld_frag(Hw + m * S128 +  0 + half * 16, 8);
  bf16x16 h1 = ld_frag(Hw + m * S128 + 32 + half * 16, 8);
  bf16x16 h2 = ld_frag(Hw + m * S128 + 64 + half * 16, 8);
  bf16x16 h3 = ld_frag(Hw + m * S128 + 96 + half * 16, 8);

#pragma unroll
  for (int nt = 0; nt < 4; nt++) {
    float bias = b2s[nt * 16 + m];
    f32x8 acc = {bias, bias, bias, bias, bias, bias, bias, bias};
    const unsigned short* bp = w2c + (nt * 16 + m) * S128 + half * 16;
    acc = __builtin_amdgcn_wmma_f32_16x16x32_bf16(false, h0, false, ld_frag(bp +  0, 8), (short)0, acc, false, false);
    acc = __builtin_amdgcn_wmma_f32_16x16x32_bf16(false, h1, false, ld_frag(bp + 32, 8), (short)0, acc, false, false);
    acc = __builtin_amdgcn_wmma_f32_16x16x32_bf16(false, h2, false, ld_frag(bp + 64, 8), (short)0, acc, false, false);
    acc = __builtin_amdgcn_wmma_f32_16x16x32_bf16(false, h3, false, ld_frag(bp + 96, 8), (short)0, acc, false, false);
#pragma unroll
    for (int r = 0; r < 8; r++)
      atomicAdd(&g[(long)dloc[r] * 64 + nt * 16 + m], acc[r]);
  }
}

// ---------------- kernel 2: g' = e*g, pooled = segment_sum(g', batch) ----------------
// batch is sorted: each thread walks 8 consecutive nodes for one channel,
// accumulating in a register and flushing one atomic per batch-segment.
__global__ void efn_pool(const float* __restrict__ g, const float* __restrict__ e,
                         const int* __restrict__ batch, float* __restrict__ pooled) {
  long gid = (long)blockIdx.x * blockDim.x + threadIdx.x;
  const long nthr = (long)(N_NODES / 8) * 64;
  if (gid >= nthr) return;
  const int c    = (int)(gid & 63);
  const int base = (int)(gid >> 6) * 8;
  int cur = batch[base];
  float s = 0.0f;
#pragma unroll
  for (int j = 0; j < 8; j++) {
    const int i = base + j;
    const int b = batch[i];
    if (b != cur) { atomicAdd(&pooled[cur * 64 + c], s); s = 0.0f; cur = b; }
    s += g[(long)i * 64 + c] * e[i];
  }
  atomicAdd(&pooled[cur * 64 + c], s);
}

// ---------------- kernel 3: local PTConv (edge MLP 256->128->1, scatter) ----------------
__global__ __launch_bounds__(128) void efn_local_conv(
    const float* __restrict__ x, const int* __restrict__ eidx,
    const int* __restrict__ batch, const float* __restrict__ pooled,
    const float* __restrict__ lW1, const float* __restrict__ lb1,
    const float* __restrict__ lW2, const float* __restrict__ lb2,
    float* __restrict__ out)
{
  __shared__ __align__(16) unsigned short w1c[128 * S256]; // lW1 col-major [n][k] bf16
  __shared__ float b1s[128];
  __shared__ float w2s[128];
  __shared__ __align__(16) unsigned short Ast[K3_WAVES][16 * S256];
  __shared__ int dstI[K3_WAVES][16];

  const int tid = threadIdx.x;
  for (int i = tid; i < 128 * 256; i += 128) { int k = i >> 7, n = i & 127; w1c[n * S256 + k] = bf16_bits(lW1[i]); }
  if (tid < 128) { b1s[tid] = lb1[tid]; w2s[tid] = lW2[tid]; }
  __syncthreads();

  const int wave = tid >> 5, lane = tid & 31, m = lane & 15, half = lane >> 4;
  const int tile = blockIdx.x * K3_WAVES + wave;
  unsigned short* Aw = Ast[wave];

  // gather: row m = [x[nd] | pooled[batch[nd]]] per half (dst half, src half)
  {
    const int e  = tile * 16 + m;
    const int nd = (half == 0) ? eidx[N_EDGES + e] : eidx[e];
    if (half == 0) dstI[wave][m] = nd;
    const int b = batch[nd];
    const float4* xr = (const float4*)(x + (long)nd * 64);
    const float4* pr = (const float4*)(pooled + b * 64);
    unsigned short* dp = Aw + m * S256 + half * 128;
#pragma unroll
    for (int j = 0; j < 16; j++) {
      float4 v = xr[j];
      *(uint2*)(dp + j * 4) = make_uint2(pk_bf16(v.x, v.y), pk_bf16(v.z, v.w));
    }
#pragma unroll
    for (int j = 0; j < 16; j++) {
      float4 v = pr[j];
      *(uint2*)(dp + 64 + j * 4) = make_uint2(pk_bf16(v.x, v.y), pk_bf16(v.z, v.w));
    }
  }
  wave_lds_fence();

  bf16x16 a[8];
#pragma unroll
  for (int s = 0; s < 8; s++) a[s] = ld_frag(Aw + m * S256 + s * 32 + half * 16, 8);

  float part[8] = {0.f, 0.f, 0.f, 0.f, 0.f, 0.f, 0.f, 0.f};
#pragma unroll
  for (int nt = 0; nt < 8; nt++) {
    float bias = b1s[nt * 16 + m];
    f32x8 acc = {bias, bias, bias, bias, bias, bias, bias, bias};
    const unsigned short* bp = w1c + (nt * 16 + m) * S256 + half * 16;
#pragma unroll
    for (int s = 0; s < 8; s++)
      acc = __builtin_amdgcn_wmma_f32_16x16x32_bf16(false, a[s], false, ld_frag(bp + s * 32, 8), (short)0, acc, false, false);
    float wv = w2s[nt * 16 + m]; // fold 128->1 projection: relu then weight, keep f32
#pragma unroll
    for (int r = 0; r < 8; r++) part[r] += fmaxf(acc[r], 0.0f) * wv;
  }
  // reduce the 16 columns held across lanes of each half
#pragma unroll
  for (int r = 0; r < 8; r++) {
    float p = part[r];
    p += __shfl_xor(p, 1, 32);
    p += __shfl_xor(p, 2, 32);
    p += __shfl_xor(p, 4, 32);
    p += __shfl_xor(p, 8, 32);
    part[r] = p;
  }
  if (m == 0) {
    float lb = lb2[0];
#pragma unroll
    for (int r = 0; r < 8; r++)
      atomicAdd(&out[dstI[wave][half * 8 + r]], part[r] + lb);
  }
}

extern "C" void kernel_launch(void* const* d_in, const int* in_sizes, int n_in,
                              void* d_out, int out_size, void* d_ws, size_t ws_size,
                              hipStream_t stream) {
  const float* x     = (const float*)d_in[0];
  const float* e     = (const float*)d_in[1];
  const int*   eidx  = (const int*)d_in[2];
  const int*   batch = (const int*)d_in[3];
  const float* gW1   = (const float*)d_in[4];
  const float* gb1   = (const float*)d_in[5];
  const float* gW2   = (const float*)d_in[6];
  const float* gb2   = (const float*)d_in[7];
  const float* lW1   = (const float*)d_in[8];
  const float* lb1   = (const float*)d_in[9];
  const float* lW2   = (const float*)d_in[10];
  const float* lb2   = (const float*)d_in[11];
  float* out    = (float*)d_out;
  float* g      = (float*)d_ws;                 // [N, 64]
  float* pooled = g + (long)N_NODES * 64;       // [64, 64]

  long ztot = (long)N_NODES * 64 + 64 * 64 + N_NODES;
  efn_zero<<<(int)((ztot + 255) / 256), 256, 0, stream>>>(g, pooled, out);
  efn_global_conv<<<N_EDGES / 16 / K1_WAVES, 256, 0, stream>>>(x, eidx, gW1, gb1, gW2, gb2, g);
  long pthr = (long)(N_NODES / 8) * 64;
  efn_pool<<<(int)((pthr + 255) / 256), 256, 0, stream>>>(g, e, batch, pooled);
  efn_local_conv<<<N_EDGES / 16 / K3_WAVES, 128, 0, stream>>>(x, eidx, batch, pooled, lW1, lb1, lW2, lb2, out);
}
